// LSTM_intensity_65403761983772
// MI455X (gfx1250) — matrix operand order repaired and checked by reading, not compiled, not run on hardware
//
#include <hip/hip_runtime.h>

// ---------------------------------------------------------------------------
// Problem constants (G grids, B batch, I input, H hidden)
// ---------------------------------------------------------------------------
constexpr int G = 512, B = 64, I = 32, H = 64;
constexpr int GATES = 4 * H;       // 256
constexpr int GH    = G * H;       // 32768
constexpr int N1 = 10 * G;         // 5120
constexpr int N2 = 5 * G;          // 2560
constexpr int N3 = G;              // 512

typedef __attribute__((ext_vector_type(2)))  float     v2f;
typedef __attribute__((ext_vector_type(4)))  float     v4f;
typedef __attribute__((ext_vector_type(8)))  float     v8f;
typedef __attribute__((ext_vector_type(4)))  int       v4i;
typedef __attribute__((ext_vector_type(8)))  _Float16  v8h;
typedef __attribute__((ext_vector_type(16))) _Float16  v16h;

// D = A(16x4,f32) * B(4x16,f32) + C, full-precision f32 WMMA (MLP path).
#define WMMA_F32(a, b, c)                                                   \
  __builtin_amdgcn_wmma_f32_16x16x4_f32(false, (a), false, (b), (short)0,   \
                                        (c), false, false)
// D = A(16x32,f16) * B(32x16,f16) + C(f32)  (LSTM recurrent path).
#define WMMA_F16(a, b, c)                                                   \
  __builtin_amdgcn_wmma_f32_16x16x32_f16(false, (a), false, (b), (short)0,  \
                                         (c), false, false)

#if __has_builtin(__builtin_amdgcn_global_load_async_to_lds_b128)
#define HAVE_ASYNC_LDS 1
// Signature (from clang diagnostics): param0 = int4 in AS(1) [__device__],
// param1 = LDS destination, imm offset, imm cpol.
#define GLBP(p) ((__attribute__((address_space(1))) v4i*)(p))
#define LDSP(p) ((__attribute__((address_space(3))) v4i*)(p))
__device__ __forceinline__ void async_wait0() {
#if __has_builtin(__builtin_amdgcn_s_wait_asynccnt)
  __builtin_amdgcn_s_wait_asynccnt(0);
#else
  asm volatile("s_wait_asynccnt 0" ::: "memory");
#endif
}
#else
#define HAVE_ASYNC_LDS 0
#endif

__device__ __forceinline__ float sigmoid_f(float x) {
  return 1.0f / (1.0f + __expf(-x));
}

// ---------------------------------------------------------------------------
// Kernel A: gates_x[g*B+b][n] = sum_k X[g*B+b][k]*W_ih[n][k] + b_ih[n]+b_hh[n]
// Fully parallel input projection for all 512 steps (f32 WMMA, exact).
// grid = 4096 blocks x 256 threads (8 waves) -> 32768 wave-tiles of 16x16.
// ---------------------------------------------------------------------------
__global__ void gates_x_kernel(const float* __restrict__ X,     // [G*B, 32]
                               const float* __restrict__ W_ih,  // [256, 32]
                               const float* __restrict__ b_ih,  // [256]
                               const float* __restrict__ b_hh,  // [256]
                               float* __restrict__ gates_x)     // [G*B, 256]
{
  const int tid = threadIdx.x, wave = tid >> 5, lane = tid & 31;
  const int half = lane >> 4, lm = lane & 15;
  const int wt = blockIdx.x * 8 + wave;   // 0..32767
  const int m0 = (wt >> 4) * 16;          // row tile of G*B
  const int n0 = (wt & 15) * 16;          // col tile of 256 gates

  v8f acc = {};
#pragma unroll
  for (int kk = 0; kk < I; kk += 4) {
    v2f a = *(const v2f*)(X + (size_t)(m0 + lm) * I + kk + 2 * half);
    v2f b = *(const v2f*)(W_ih + (size_t)(n0 + lm) * I + kk + 2 * half);
    acc = WMMA_F32(a, b, acc);
  }
  const int col = n0 + lm;
  const float bias = b_ih[col] + b_hh[col];
#pragma unroll
  for (int r = 0; r < 8; ++r) {
    const int row = m0 + r + 8 * half;
    gates_x[(size_t)row * GATES + col] = acc[r] + bias;
  }
}

// ---------------------------------------------------------------------------
// Kernel B: sequential LSTM scan (the latency-critical path).
// 4 blocks own 16 independent batch rows each; 4 waves per block; wave j owns
// hidden columns [16j,16j+16) for ALL FOUR gates so the cell update stays in
// registers. Recurrent matmul runs in f16 WMMA (16x16x32, f32 accum, state in
// f32). W_hh B-operands register-resident as f16. h crosses waves via a
// double-buffered f16 LDS tile -> ONE barrier per step. gates_x loads for
// step g+1 are software-pipelined behind step g's WMMAs.
// ---------------------------------------------------------------------------
__global__ void lstm_scan_kernel(const float* __restrict__ gates_x, // [G,B,256]
                                 const float* __restrict__ h0,      // [B,64]
                                 const float* __restrict__ c0,      // [B,64]
                                 const float* __restrict__ W_hh,    // [256,64]
                                 float* __restrict__ out_all)       // [B, G*64]
{
  __shared__ _Float16 hbuf[2][16 * 72];   // [batch 16][K 64] f16, stride 72
  const int tid = threadIdx.x, wave = tid >> 5, lane = tid & 31;
  const int half = lane >> 4, lm = lane & 15;
  const int bs = blockIdx.x * 16;  // batch slice
  const int j = wave;              // hidden-column tile

  // --- W_hh -> register-resident f16 B operands ---------------------------
  // B(32x16) layout: VGPR v holds K = 32*w + 8*half + 8*(v>=4) + {2v,2v+1}, N = lm.
  v16h bw[4][2];
#pragma unroll
  for (int gate = 0; gate < 4; ++gate) {
    const float* wrow = W_hh + (size_t)(gate * 64 + j * 16 + lm) * H;
#pragma unroll
    for (int w = 0; w < 2; ++w) {
      v16h bb;
#pragma unroll
      for (int v = 0; v < 8; ++v) {
        const int K0 = 32 * w + 8 * half + 2 * v + (v >= 4 ? 8 : 0);
        v2f t = *(const v2f*)(wrow + K0);
        bb[2 * v]     = (_Float16)t.x;
        bb[2 * v + 1] = (_Float16)t.y;
      }
      bw[gate][w] = bb;
    }
  }

  // --- init h (f16 LDS) and c (f32 regs, C/D layout) -----------------------
  for (int v = tid; v < 16 * 64; v += 128) {
    const int m = v >> 6, col = v & 63;
    hbuf[0][m * 72 + col] = (_Float16)h0[(size_t)(bs + m) * H + col];
  }
  float c[8];
#pragma unroll
  for (int r = 0; r < 8; ++r)
    c[r] = c0[(size_t)(bs + r + 8 * half) * H + j * 16 + lm];

  // --- gates_x prefetch pipeline -------------------------------------------
  float gxc[4][8], gxn[4][8];
  {
    const float* gx = gates_x + (size_t)bs * GATES;
#pragma unroll
    for (int gate = 0; gate < 4; ++gate)
#pragma unroll
      for (int r = 0; r < 8; ++r)
        gxc[gate][r] = gx[(size_t)(r + 8 * half) * GATES + gate * 64 + j * 16 + lm];
  }
  __syncthreads();

  int pb = 0;
  for (int g = 0; g < G; ++g) {
    if (g + 1 < G) {  // prefetch next step's input gates (hidden behind WMMAs)
      const float* gx = gates_x + ((size_t)(g + 1) * B + bs) * GATES;
#pragma unroll
      for (int gate = 0; gate < 4; ++gate)
#pragma unroll
        for (int r = 0; r < 8; ++r)
          gxn[gate][r] = gx[(size_t)(r + 8 * half) * GATES + gate * 64 + j * 16 + lm];
    }
    // A operand (16x32 f16): lane holds row m=lm, K runs {8h..8h+7, 16+8h..}
    const _Float16* hb = hbuf[pb];
    v8h lo0 = *(const v8h*)&hb[lm * 72 +      8 * half];
    v8h hi0 = *(const v8h*)&hb[lm * 72 + 16 + 8 * half];
    v8h lo1 = *(const v8h*)&hb[lm * 72 + 32 + 8 * half];
    v8h hi1 = *(const v8h*)&hb[lm * 72 + 48 + 8 * half];
    v16h a0 = __builtin_shufflevector(lo0, hi0, 0,1,2,3,4,5,6,7,8,9,10,11,12,13,14,15);
    v16h a1 = __builtin_shufflevector(lo1, hi1, 0,1,2,3,4,5,6,7,8,9,10,11,12,13,14,15);

    v8f zero = {};
    v8f acc[4];
#pragma unroll
    for (int gate = 0; gate < 4; ++gate) {
      acc[gate] = WMMA_F16(a0, bw[gate][0], zero);
      acc[gate] = WMMA_F16(a1, bw[gate][1], acc[gate]);
    }

    float hv[8];
#pragma unroll
    for (int r = 0; r < 8; ++r) {
      const float iv = sigmoid_f(acc[0][r] + gxc[0][r]);
      const float fv = sigmoid_f(acc[1][r] + gxc[1][r]);
      const float gv = tanhf(acc[2][r] + gxc[2][r]);
      const float ov = sigmoid_f(acc[3][r] + gxc[3][r]);
      c[r] = fv * c[r] + iv * gv;
      hv[r] = ov * tanhf(c[r]);
      out_all[(size_t)(bs + r + 8 * half) * GH + g * H + j * 16 + lm] = hv[r];
    }
    _Float16* hn = hbuf[pb ^ 1];
#pragma unroll
    for (int r = 0; r < 8; ++r)
      hn[(r + 8 * half) * 72 + j * 16 + lm] = (_Float16)hv[r];
    __syncthreads();   // single barrier: writes(pb^1) visible, reads(pb) done
    pb ^= 1;
#pragma unroll
    for (int gate = 0; gate < 4; ++gate)
#pragma unroll
      for (int r = 0; r < 8; ++r)
        gxc[gate][r] = gxn[gate][r];
  }
}

// ---------------------------------------------------------------------------
// Memory-bound GEMM: out[64,N] = A[64,K] @ W[N,K]^T + bias (opt sigmoid).
// 256-thread block computes a 64x64 strip. K chunked by 32 through a
// DOUBLE-BUFFERED LDS stage (padded stride 36 -> conflict-free b64 operand
// reads; 16B-aligned b128 transfers). Staging uses the CDNA5 async
// global->LDS path (ASYNCcnt) when available so the 671MB W1 stream overlaps
// the f32 WMMAs with a single barrier per chunk; otherwise a register-staged
// software pipeline. f32 WMMA: weight bytes are f32 regardless, HBM-bound.
// ---------------------------------------------------------------------------
__global__ void gemm64_kernel(const float* __restrict__ A,    // [64, K]
                              const float* __restrict__ W,    // [N, K]
                              const float* __restrict__ bias, // [N]
                              float* __restrict__ out,        // [64, N]
                              int K, int N, int do_sigmoid)
{
  constexpr int KC = 32;    // k-chunk
  constexpr int LST = 36;   // padded LDS row stride (floats); 144B, 16B-aligned
  __shared__ float As[2][64 * LST];
  __shared__ float Ws[2][64 * LST];

  const int tid = threadIdx.x, wave = tid >> 5, lane = tid & 31;
  const int half = lane >> 4, lm = lane & 15;
  const int N0 = blockIdx.x * 64;
  const int m0 = (wave & 3) * 16;        // M tile
  const int p  = wave >> 2;              // N pair
  const int nt0 = p * 32, nt1 = p * 32 + 16;

  // Per-thread staging map: 2 x b128 per array per chunk.
  // idx = tid + 256*i in [0,512): row m = idx>>3, quad q = idx&7.
  const int sm0 = tid >> 3,          sq0 = tid & 7;
  const int sm1 = (tid + 256) >> 3,  sq1 = tid & 7;

  v8f acc0 = {}, acc1 = {};

  // ---- prologue: stage chunk 0 into buffer 0 ------------------------------
#if HAVE_ASYNC_LDS
  __builtin_amdgcn_global_load_async_to_lds_b128(
      GLBP(A + (size_t)sm0 * K + sq0 * 4), LDSP(&As[0][sm0 * LST + sq0 * 4]), 0, 0);
  __builtin_amdgcn_global_load_async_to_lds_b128(
      GLBP(A + (size_t)sm1 * K + sq1 * 4), LDSP(&As[0][sm1 * LST + sq1 * 4]), 0, 0);
  __builtin_amdgcn_global_load_async_to_lds_b128(
      GLBP(W + (size_t)(N0 + sm0) * K + sq0 * 4), LDSP(&Ws[0][sm0 * LST + sq0 * 4]), 0, 0);
  __builtin_amdgcn_global_load_async_to_lds_b128(
      GLBP(W + (size_t)(N0 + sm1) * K + sq1 * 4), LDSP(&Ws[0][sm1 * LST + sq1 * 4]), 0, 0);
  async_wait0();
#else
  {
    v4f ra0 = *(const v4f*)(A + (size_t)sm0 * K + sq0 * 4);
    v4f ra1 = *(const v4f*)(A + (size_t)sm1 * K + sq1 * 4);
    v4f rw0 = *(const v4f*)(W + (size_t)(N0 + sm0) * K + sq0 * 4);
    v4f rw1 = *(const v4f*)(W + (size_t)(N0 + sm1) * K + sq1 * 4);
    *(v4f*)&As[0][sm0 * LST + sq0 * 4] = ra0;
    *(v4f*)&As[0][sm1 * LST + sq1 * 4] = ra1;
    *(v4f*)&Ws[0][sm0 * LST + sq0 * 4] = rw0;
    *(v4f*)&Ws[0][sm1 * LST + sq1 * 4] = rw1;
  }
#endif
  __syncthreads();

  int buf = 0;
  for (int kc = 0; kc < K; kc += KC) {
    const bool more = (kc + KC) < K;
    const int kn = kc + KC;
#if HAVE_ASYNC_LDS
    if (more) {  // issue next chunk into the other buffer (reads of it done)
      __builtin_amdgcn_global_load_async_to_lds_b128(
          GLBP(A + (size_t)sm0 * K + kn + sq0 * 4),
          LDSP(&As[buf ^ 1][sm0 * LST + sq0 * 4]), 0, 0);
      __builtin_amdgcn_global_load_async_to_lds_b128(
          GLBP(A + (size_t)sm1 * K + kn + sq1 * 4),
          LDSP(&As[buf ^ 1][sm1 * LST + sq1 * 4]), 0, 0);
      __builtin_amdgcn_global_load_async_to_lds_b128(
          GLBP(W + (size_t)(N0 + sm0) * K + kn + sq0 * 4),
          LDSP(&Ws[buf ^ 1][sm0 * LST + sq0 * 4]), 0, 0);
      __builtin_amdgcn_global_load_async_to_lds_b128(
          GLBP(W + (size_t)(N0 + sm1) * K + kn + sq1 * 4),
          LDSP(&Ws[buf ^ 1][sm1 * LST + sq1 * 4]), 0, 0);
    }
#else
    v4f ra0, ra1, rw0, rw1;
    if (more) {  // global loads issued now, consumed after compute
      ra0 = *(const v4f*)(A + (size_t)sm0 * K + kn + sq0 * 4);
      ra1 = *(const v4f*)(A + (size_t)sm1 * K + kn + sq1 * 4);
      rw0 = *(const v4f*)(W + (size_t)(N0 + sm0) * K + kn + sq0 * 4);
      rw1 = *(const v4f*)(W + (size_t)(N0 + sm1) * K + kn + sq1 * 4);
    }
#endif
    if (kc + 2 * KC < K)  // far L2 prefetch of the W stream
      __builtin_prefetch(W + (size_t)(N0 + (tid >> 2)) * K + kc + 2 * KC + (tid & 3) * 8, 0, 1);

#pragma unroll
    for (int kq = 0; kq < 8; ++kq) {
      v2f a  = *(const v2f*)&As[buf][(m0  + lm) * LST + kq * 4 + 2 * half];
      v2f b0 = *(const v2f*)&Ws[buf][(nt0 + lm) * LST + kq * 4 + 2 * half];
      v2f b1 = *(const v2f*)&Ws[buf][(nt1 + lm) * LST + kq * 4 + 2 * half];
      acc0 = WMMA_F32(a, b0, acc0);
      acc1 = WMMA_F32(a, b1, acc1);
    }

#if HAVE_ASYNC_LDS
    if (more) async_wait0();
#else
    if (more) {
      *(v4f*)&As[buf ^ 1][sm0 * LST + sq0 * 4] = ra0;
      *(v4f*)&As[buf ^ 1][sm1 * LST + sq1 * 4] = ra1;
      *(v4f*)&Ws[buf ^ 1][sm0 * LST + sq0 * 4] = rw0;
      *(v4f*)&Ws[buf ^ 1][sm1 * LST + sq1 * 4] = rw1;
    }
#endif
    __syncthreads();
    buf ^= 1;
  }

#pragma unroll
  for (int r = 0; r < 8; ++r) {
    const int m  = m0 + r + 8 * half;
    const int n0 = N0 + nt0 + lm;
    const int n1 = N0 + nt1 + lm;
    float v0 = acc0[r] + bias[n0];
    float v1 = acc1[r] + bias[n1];
    if (do_sigmoid) { v0 = sigmoid_f(v0); v1 = sigmoid_f(v1); }
    out[(size_t)m * N + n0] = v0;
    out[(size_t)m * N + n1] = v1;
  }
}

// ---------------------------------------------------------------------------
// Host launcher
// ---------------------------------------------------------------------------
extern "C" void kernel_launch(void* const* d_in, const int* in_sizes, int n_in,
                              void* d_out, int out_size, void* d_ws, size_t ws_size,
                              hipStream_t stream)
{
  const float* input_data = (const float*)d_in[0];   // [G,B,I]
  const float* h0   = (const float*)d_in[1];         // [B,H]
  const float* c0   = (const float*)d_in[2];         // [B,H]
  const float* W_ih = (const float*)d_in[3];         // [4H,I]
  const float* W_hh = (const float*)d_in[4];         // [4H,H]
  const float* b_ih = (const float*)d_in[5];         // [4H]
  const float* b_hh = (const float*)d_in[6];         // [4H]
  const float* W1   = (const float*)d_in[7];         // [10G, H*G]
  const float* b1   = (const float*)d_in[8];
  const float* W2   = (const float*)d_in[9];         // [5G, 10G]
  const float* b2   = (const float*)d_in[10];
  const float* W3   = (const float*)d_in[11];        // [G, 5G]
  const float* b3   = (const float*)d_in[12];

  float* gates_x = (float*)d_ws;                       // G*B*256  = 33.55 MB
  float* out_all = gates_x + (size_t)G * B * GATES;    // B*G*H    =  8.39 MB
  float* x1      = out_all + (size_t)B * GH;           // B*N1     =  1.31 MB
  float* x2      = x1 + (size_t)B * N1;                // B*N2     =  0.66 MB
  float* out     = (float*)d_out;                      // [B, G]

  gates_x_kernel<<<4096, 256, 0, stream>>>(input_data, W_ih, b_ih, b_hh, gates_x);
  lstm_scan_kernel<<<B / 16, 128, 0, stream>>>(gates_x, h0, c0, W_hh, out_all);
  gemm64_kernel<<<N1 / 64, 256, 0, stream>>>(out_all, W1, b1, x1, GH, N1, 0);
  gemm64_kernel<<<N2 / 64, 256, 0, stream>>>(x1, W2, b2, x2, N1, N2, 0);
  gemm64_kernel<<<N3 / 64, 256, 0, stream>>>(x2, W3, b3, out, N2, N3, 1);
}
